// TransducerBeamSearcher_11269994185110
// MI455X (gfx1250) — matrix operand order repaired and checked by reading, not compiled, not run on hardware
//
#include <hip/hip_runtime.h>
#include <hip/hip_bf16.h>

// ---------------- problem constants ----------------
#define B_   32
#define T_   500
#define F_   640
#define H_   640
#define J_   640
#define V_   1024
#define G4_  2560          // 4*H
#define KT_  20            // 640 / 32 (K tiles per GEMM)
#define NWG  32            // persistent workgroups
#define TPB  256           // threads per block (8 wave32)

typedef __attribute__((ext_vector_type(16))) __bf16        v16bf;
typedef __attribute__((ext_vector_type(8)))  float         v8f;
typedef __attribute__((ext_vector_type(4)))  unsigned int  u32x4;
typedef __attribute__((ext_vector_type(4)))  float         f32x4;

union Frag { v16bf v; u32x4 q[2]; unsigned int u[8]; };

// fp32 -> bf16 (round-to-nearest-even)
__device__ inline unsigned short f2bf(float x) {
  unsigned int u = __builtin_bit_cast(unsigned int, x);
  u += 0x7FFFu + ((u >> 16) & 1u);
  return (unsigned short)(u >> 16);
}
__device__ inline unsigned int pk2(float a, float b) {
  return (unsigned int)f2bf(a) | ((unsigned int)f2bf(b) << 16);
}
__device__ inline float sigmoidf_(float x) { return 1.0f / (1.0f + __expf(-x)); }

__device__ inline v8f wmma_bf16(const Frag& a, const Frag& b, v8f c) {
  return __builtin_amdgcn_wmma_f32_16x16x32_bf16(false, a.v, false, b.v,
                                                 (short)0, c, false, false);
}

// A fragment (16x32 bf16) from a row-major bf16 matrix. ISA A layout for a
// 16-bit 16x32 tile decomposes, per lane, into two contiguous 16B chunks:
//   lane L: row M = L%16, half h = L/16
//   u[0..3] <- K = 8h .. 8h+7       (bytes [16h, 16h+16) of the 64B K-block)
//   u[4..7] <- K = 16+8h .. 16+8h+7 (bytes [32+16h, 32+16h+16))
__device__ inline Frag load_a_bf16(const unsigned short* base, int row, int ld,
                                   int kt, int lane) {
  int half = lane >> 4;
  const char* p = (const char*)(base + (size_t)row * ld + kt * 32) + half * 16;
  Frag f;
  f.q[0] = *(const u32x4*)(p);
  f.q[1] = *(const u32x4*)(p + 32);
  return f;
}

// B fragment from a pre-swizzled weight buffer: [(nt*KT + kt)*32 + lane] * 32B
__device__ inline Frag load_b_packed(const unsigned int* packed, int nt, int kt,
                                     int kts, int lane) {
  const unsigned int* p = packed + ((size_t)(nt * kts + kt) * 32 + lane) * 8;
  Frag f;
  f.q[0] = *(const u32x4*)(p);
  f.q[1] = *(const u32x4*)(p + 4);
  return f;
}

// Monotonic grid-wide barrier (persistent kernel, all WGs co-resident)
__device__ inline void grid_barrier(unsigned int* cnt, unsigned int target) {
  __threadfence();
  __syncthreads();
  if (threadIdx.x == 0) {
    __hip_atomic_fetch_add(cnt, 1u, __ATOMIC_ACQ_REL, __HIP_MEMORY_SCOPE_AGENT);
    while (__hip_atomic_load(cnt, __ATOMIC_ACQUIRE, __HIP_MEMORY_SCOPE_AGENT) < target)
      __builtin_amdgcn_s_sleep(2);
  }
  __syncthreads();
  __threadfence();
}

// ---------------- weight packing: fp32 row-major (K x N) -> bf16 WMMA B frags
// B layout (32x16 per tile): lane L holds col N = L%16; elements e=0..7 are
// K = 8*(L/16) + e, e=8..15 are K = 16 + 8*(L/16) + (e-8) within the 32-K block.
__global__ void pack_b_kernel(const float* __restrict__ W,
                              unsigned int* __restrict__ out, int K, int N) {
  int ktiles = K >> 5, ntiles = N >> 4;
  int total = ntiles * ktiles * 32;
  int idx = blockIdx.x * blockDim.x + threadIdx.x;
  if (idx >= total) return;
  int lane = idx & 31;
  int tile = idx >> 5;
  int kt = tile % ktiles;
  int nt = tile / ktiles;
  int n = nt * 16 + (lane & 15);
  int kbase = kt * 32 + (lane >> 4) * 8;
  unsigned int* dst = out + (size_t)idx * 8;
#pragma unroll
  for (int i = 0; i < 8; ++i) {
    float a = W[(size_t)(kbase + 2 * i + (i >> 2) * 8)     * N + n];
    float b = W[(size_t)(kbase + 2 * i + (i >> 2) * 8 + 1) * N + n];
    dst[i] = pk2(a, b);
  }
}

// ---------------- hoisted GEMM: encpre = tn_output @ W_enc + b_j ------------
// 16000x640 @ 640x640. One wave per 16x16 output tile, fp32->bf16 on the fly.
__global__ void encproj_kernel(const float* __restrict__ tn,
                               const unsigned int* __restrict__ pWenc,
                               const float* __restrict__ bj,
                               float* __restrict__ encpre) {
  int wave = threadIdx.x >> 5, lane = threadIdx.x & 31;
  int tile = blockIdx.x * 8 + wave;          // 5000 blocks * 8 = 40000 tiles
  int mt = tile / 40, nt = tile % 40;
  int half = lane >> 4;
  int row = mt * 16 + (lane & 15);
  v8f acc = {0.f, 0.f, 0.f, 0.f, 0.f, 0.f, 0.f, 0.f};
  for (int kt = 0; kt < KT_; ++kt) {
    const float* p = tn + (size_t)row * F_ + kt * 32;
    f32x4 a0 = *(const f32x4*)(p + half * 8);
    f32x4 a1 = *(const f32x4*)(p + half * 8 + 4);
    f32x4 a2 = *(const f32x4*)(p + 16 + half * 8);
    f32x4 a3 = *(const f32x4*)(p + 16 + half * 8 + 4);
    Frag fa;
    fa.u[0] = pk2(a0.x, a0.y); fa.u[1] = pk2(a0.z, a0.w);
    fa.u[2] = pk2(a1.x, a1.y); fa.u[3] = pk2(a1.z, a1.w);
    fa.u[4] = pk2(a2.x, a2.y); fa.u[5] = pk2(a2.z, a2.w);
    fa.u[6] = pk2(a3.x, a3.y); fa.u[7] = pk2(a3.z, a3.w);
    Frag fb = load_b_packed(pWenc, nt, kt, KT_, lane);
    acc = wmma_bf16(fa, fb, acc);
  }
  int col = nt * 16 + (lane & 15);
  float bias = bj[col];
#pragma unroll
  for (int v = 0; v < 8; ++v) {
    int r = mt * 16 + v + 8 * half;          // D layout: M = v + 8*(lane/16)
    encpre[(size_t)r * J_ + col] = acc[v] + bias;
  }
}

// ---------------- scratch init (runs every launch; ws is poisoned) ----------
__global__ void init_kernel(unsigned short* hbf, unsigned short* gbf,
                            float* cbuf, int* last_tok, unsigned int* counter) {
  int i = blockIdx.x * blockDim.x + threadIdx.x;
  if (i < B_ * H_) { hbf[i] = 0; gbf[i] = 0; cbuf[i] = 0.f; }
  if (i < 64) last_tok[i] = 0;
  if (i == 0) *counter = 0;
}

// ---------------- persistent greedy-decode kernel ---------------------------
__launch_bounds__(TPB, 1)
__global__ void decode_kernel(
    const float* __restrict__ E, const float* __restrict__ b_lstm,
    const float* __restrict__ b_out,
    const unsigned int* __restrict__ pWdec, const unsigned int* __restrict__ pWout,
    const unsigned int* __restrict__ pWih,  const unsigned int* __restrict__ pWhh,
    const float* __restrict__ encpre,
    unsigned short* __restrict__ abf, float* __restrict__ logits,
    float* __restrict__ gates,
    unsigned short* __restrict__ hbf, unsigned short* __restrict__ gbf,
    float* __restrict__ cbuf, unsigned short* __restrict__ xbf_all,
    int* __restrict__ last_tok, unsigned int* __restrict__ counter,
    float* __restrict__ out) {
  __shared__ int   s_tok[32];
  __shared__ float s_lp[32];
  __shared__ int   s_newtok[32];
  __shared__ int   s_emit[32];
  __shared__ float s_scores[32];

  const int tid = threadIdx.x;
  const int wave = tid >> 5, lane = tid & 31;
  const int wg = blockIdx.x;
  const int gwave = wg * 8 + wave;                 // 256 global waves
  unsigned short* xbf = xbf_all + (size_t)wg * (B_ * H_);  // WG-private
  unsigned int epoch = 0;

  if (wg == 0 && tid < 32) s_scores[tid] = 0.f;

  for (int t = -1; t < T_; ++t) {
    if (t >= 0) {
      // ---- Phase A: a = tanh(encpre[:,t,:] + g @ W_dec), stored bf16 ----
      for (int task = gwave; task < 2 * (J_ / 16); task += NWG * 8) {
        int jt = task >> 1, mh = task & 1;
        int row = mh * 16 + (lane & 15);
        v8f acc = {0.f, 0.f, 0.f, 0.f, 0.f, 0.f, 0.f, 0.f};
        for (int kt = 0; kt < KT_; ++kt) {
          Frag fa = load_a_bf16(gbf, row, H_, kt, lane);
          Frag fb = load_b_packed(pWdec, jt, kt, KT_, lane);
          acc = wmma_bf16(fa, fb, acc);
        }
        int col = jt * 16 + (lane & 15);
        int half = lane >> 4;
#pragma unroll
        for (int v = 0; v < 8; ++v) {
          int b = mh * 16 + v + 8 * half;
          float val = tanhf(acc[v] + encpre[((size_t)b * T_ + t) * J_ + col]);
          abf[b * J_ + col] = f2bf(val);
        }
      }
      ++epoch; grid_barrier(counter, epoch * NWG);

      // ---- Phase B: logits = a @ W_out + b_out ----
      for (int task = gwave; task < 2 * (V_ / 16); task += NWG * 8) {
        int vt = task >> 1, mh = task & 1;
        int row = mh * 16 + (lane & 15);
        v8f acc = {0.f, 0.f, 0.f, 0.f, 0.f, 0.f, 0.f, 0.f};
        for (int kt = 0; kt < KT_; ++kt) {
          Frag fa = load_a_bf16(abf, row, J_, kt, lane);
          Frag fb = load_b_packed(pWout, vt, kt, KT_, lane);
          acc = wmma_bf16(fa, fb, acc);
        }
        int col = vt * 16 + (lane & 15);
        int half = lane >> 4;
        float bias = b_out[col];
#pragma unroll
        for (int v = 0; v < 8; ++v)
          logits[(size_t)(mh * 16 + v + 8 * half) * V_ + col] = acc[v] + bias;
      }
      ++epoch; grid_barrier(counter, epoch * NWG);

      // ---- Redundant per-WG argmax + log-softmax (lp = -log(sum exp)) ----
      for (int r = wave; r < 32; r += 8) {
        const float* lr = logits + (size_t)r * V_;
        float m = -3.4e38f; int arg = 0;
        for (int i = lane; i < V_; i += 32) {
          float v = lr[i];
          if (v > m) { m = v; arg = i; }
        }
#pragma unroll
        for (int off = 16; off > 0; off >>= 1) {
          float m2 = __shfl_xor(m, off, 32);
          int   a2 = __shfl_xor(arg, off, 32);
          if (m2 > m || (m2 == m && a2 < arg)) { m = m2; arg = a2; }
        }
        float s = 0.f;
        for (int i = lane; i < V_; i += 32) s += __expf(lr[i] - m);
#pragma unroll
        for (int off = 16; off > 0; off >>= 1) s += __shfl_xor(s, off, 32);
        if (lane == 0) { s_tok[r] = arg; s_lp[r] = -__logf(s); }
      }
      __syncthreads();
      if (tid < 32) {
        int tok = s_tok[tid];
        int em = (tok != 0);
        int lt = last_tok[(t & 1) * 32 + tid];      // parity double-buffer
        int nt2 = em ? tok : lt;
        s_newtok[tid] = nt2;
        s_emit[tid] = em;
        if (wg == 0) {                               // single writer
          if (em) s_scores[tid] += s_lp[tid];
          last_tok[((t & 1) ^ 1) * 32 + tid] = nt2;
          out[(size_t)tid * T_ + t] = em ? (float)tok : 0.0f;
        }
      }
      __syncthreads();
    } else {
      if (tid < 32) { s_newtok[tid] = 0; s_emit[tid] = 1; }  // init LSTM step
      __syncthreads();
    }

    // ---- per-WG private gather: x = bf16(E[new_tok]) ----
    for (int i = tid; i < B_ * H_; i += TPB) {
      int b = i / H_;
      xbf[i] = f2bf(E[(size_t)s_newtok[b] * H_ + (i % H_)]);
    }
    __syncthreads();

    // ---- Phase C: gates = x @ W_ih + h @ W_hh + b_lstm ----
    for (int task = gwave; task < 2 * (G4_ / 16); task += NWG * 8) {
      int gt = task >> 1, mh = task & 1;
      int row = mh * 16 + (lane & 15);
      v8f acc = {0.f, 0.f, 0.f, 0.f, 0.f, 0.f, 0.f, 0.f};
      for (int kt = 0; kt < KT_; ++kt) {
        Frag fa = load_a_bf16(xbf, row, H_, kt, lane);
        Frag fb = load_b_packed(pWih, gt, kt, KT_, lane);
        acc = wmma_bf16(fa, fb, acc);
        Frag fa2 = load_a_bf16(hbf, row, H_, kt, lane);
        Frag fb2 = load_b_packed(pWhh, gt, kt, KT_, lane);
        acc = wmma_bf16(fa2, fb2, acc);
      }
      int col = gt * 16 + (lane & 15);
      int half = lane >> 4;
      float bias = b_lstm[col];
#pragma unroll
      for (int v = 0; v < 8; ++v)
        gates[(size_t)(mh * 16 + v + 8 * half) * G4_ + col] = acc[v] + bias;
    }
    ++epoch; grid_barrier(counter, epoch * NWG);

    // ---- Phase E: conditional LSTM state update (partitioned) ----
    for (int i = wg * TPB + tid; i < B_ * H_; i += NWG * TPB) {
      int b = i / H_;
      if (s_emit[b]) {
        int j = i % H_;
        const float* gr = gates + (size_t)b * G4_;
        float ig = gr[j], fg = gr[j + H_], gg = gr[j + 2 * H_], og = gr[j + 3 * H_];
        float c2 = sigmoidf_(fg) * cbuf[i] + sigmoidf_(ig) * tanhf(gg);
        float h2 = sigmoidf_(og) * tanhf(c2);
        cbuf[i] = c2;
        unsigned short hb = f2bf(h2);
        hbf[i] = hb;
        gbf[i] = hb;   // g = h2 where emit (unconditional at t==-1: emit forced)
      }
    }
    ++epoch; grid_barrier(counter, epoch * NWG);
  }

  if (wg == 0) {
    __syncthreads();
    if (tid == 0) {
      float s = 0.f;
      for (int b = 0; b < 32; ++b) s += __expf(s_scores[b]);
      out[(size_t)B_ * T_] = s / (float)B_;   // exp(scores).mean()
    }
  }
}

// ---------------- host side -------------------------------------------------
extern "C" void kernel_launch(void* const* d_in, const int* in_sizes, int n_in,
                              void* d_out, int out_size, void* d_ws, size_t ws_size,
                              hipStream_t stream) {
  const float* tn     = (const float*)d_in[0];
  const float* E      = (const float*)d_in[1];
  const float* W_ih   = (const float*)d_in[2];
  const float* W_hh   = (const float*)d_in[3];
  const float* b_lstm = (const float*)d_in[4];
  const float* W_enc  = (const float*)d_in[5];
  const float* W_dec  = (const float*)d_in[6];
  const float* b_j    = (const float*)d_in[7];
  const float* W_out  = (const float*)d_in[8];
  const float* b_out  = (const float*)d_in[9];
  float* out = (float*)d_out;

  char* p = (char*)d_ws;
  auto alloc = [&](size_t bytes) -> char* {
    char* r = p;
    p += (bytes + 255) & ~(size_t)255;
    return r;
  };
  float*          encpre  = (float*)alloc((size_t)B_ * T_ * J_ * 4);     // 41 MB
  unsigned int*   pWenc   = (unsigned int*)alloc((size_t)40 * 20 * 32 * 32);
  unsigned int*   pWdec   = (unsigned int*)alloc((size_t)40 * 20 * 32 * 32);
  unsigned int*   pWout   = (unsigned int*)alloc((size_t)64 * 20 * 32 * 32);
  unsigned int*   pWih    = (unsigned int*)alloc((size_t)160 * 20 * 32 * 32);
  unsigned int*   pWhh    = (unsigned int*)alloc((size_t)160 * 20 * 32 * 32);
  unsigned short* abf     = (unsigned short*)alloc((size_t)B_ * J_ * 2);
  float*          logits  = (float*)alloc((size_t)B_ * V_ * 4);
  float*          gates   = (float*)alloc((size_t)B_ * G4_ * 4);
  unsigned short* hbf     = (unsigned short*)alloc((size_t)B_ * H_ * 2);
  unsigned short* gbf     = (unsigned short*)alloc((size_t)B_ * H_ * 2);
  float*          cbuf    = (float*)alloc((size_t)B_ * H_ * 4);
  unsigned short* xbf     = (unsigned short*)alloc((size_t)NWG * B_ * H_ * 2);
  int*            ltok    = (int*)alloc(64 * 4);
  unsigned int*   counter = (unsigned int*)alloc(256);

  auto packLaunch = [&](const float* W, unsigned int* dst, int K, int N) {
    int total = (N / 16) * (K / 32) * 32;
    pack_b_kernel<<<(total + 255) / 256, 256, 0, stream>>>(W, dst, K, N);
  };
  packLaunch(W_enc, pWenc, 640, 640);
  packLaunch(W_dec, pWdec, 640, 640);
  packLaunch(W_out, pWout, 640, 1024);
  packLaunch(W_ih,  pWih,  640, 2560);
  packLaunch(W_hh,  pWhh,  640, 2560);

  encproj_kernel<<<5000, 256, 0, stream>>>(tn, pWenc, b_j, encpre);
  init_kernel<<<(B_ * H_ + 255) / 256, 256, 0, stream>>>(hbf, gbf, cbuf, ltok, counter);
  decode_kernel<<<NWG, TPB, 0, stream>>>(E, b_lstm, b_out, pWdec, pWout, pWih, pWhh,
                                         encpre, abf, logits, gates, hbf, gbf, cbuf,
                                         xbf, ltok, counter, out);
}
